// GraphLearningModule_60756607369732
// MI455X (gfx1250) — compile-verified, matrix operand order checked by reading, not run on hardware
//
#include <hip/hip_runtime.h>
#include <hip/hip_bf16.h>
#include <math.h>

// ---------------- problem constants (match reference) ----------------
#define BATCH 8
#define NNODE 2048           // N
#define CDIM  256            // C == E
#define TOPK  32
#define SLOPE 0.2f
#define ROWS  (BATCH * NNODE)            // 16384
#define BNN   ((size_t)BATCH * NNODE * NNODE)   // 33554432

// ---------------- WMMA types (probe-confirmed) ----------------
typedef __attribute__((ext_vector_type(16))) _Float16 v16h;
typedef __attribute__((ext_vector_type(8)))  _Float16 v8h;
typedef __attribute__((ext_vector_type(8)))  float    v8f;

// Load a 16x32 f16 A-style fragment (also valid for B = 32x16 loaded from the
// row-major "transposed" operand: per-lane mapping is identical, row/col = lane&15).
// ISA 7.12.2: lane<16 -> K = {kb..kb+7, 16+kb..16+kb+7} with kb = (lane>>4)*8.
__device__ inline v16h load_frag16(const _Float16* __restrict__ base, int ld, int lane) {
    int row = lane & 15;
    int kb  = (lane >> 4) << 3;
    const _Float16* p = base + (size_t)row * ld + kb;
    v8h lo = *(const v8h*)(p);
    v8h hi = *(const v8h*)(p + 16);
    v16h f;
#pragma unroll
    for (int i = 0; i < 8; ++i) { f[i] = lo[i]; f[i + 8] = hi[i]; }
    return f;
}

// ---------------- prep: f32 -> f16 convert of x ----------------
__global__ void cvt_x_kernel(const float* __restrict__ x, _Float16* __restrict__ x16, int n) {
    int i = blockIdx.x * blockDim.x + threadIdx.x;
    if (i < n) x16[i] = (_Float16)x[i];
}

// ---------------- prep: transpose+convert Wq, Wk -> [E x C] f16 ----------------
__global__ void cvt_wT_kernel(const float* __restrict__ Wq, const float* __restrict__ Wk,
                              _Float16* __restrict__ wqT, _Float16* __restrict__ wkT) {
    int i = blockIdx.x * blockDim.x + threadIdx.x;      // 0 .. 2*65536-1
    int which = i >> 16;
    int e = (i >> 8) & 255;
    int c = i & 255;
    if (which == 0) wqT[e * CDIM + c] = (_Float16)Wq[c * CDIM + e];
    else            wkT[e * CDIM + c] = (_Float16)Wk[c * CDIM + e];
}

// ---------------- per-batch column sums of x (for the loss) ----------------
__global__ void colsum_kernel(const float* __restrict__ x, float* __restrict__ Sb) {
    int c = threadIdx.x;               // 0..255
    int b = blockIdx.x;                // 0..7
    const float* xb = x + (size_t)b * NNODE * CDIM;
    float s = 0.f;
    for (int n = 0; n < NNODE; ++n) s += xb[(size_t)n * CDIM + c];
    Sb[b * CDIM + c] = s;
}

// ---------------- projection GEMM: q16/k16 = f16(x @ W + bias) ----------------
// grid = (E/16, ROWS/16/4, 2), block = 128 (4 waves, one 16x16 tile each)
__global__ void proj_kernel(const _Float16* __restrict__ x16,
                            const _Float16* __restrict__ wqT, const _Float16* __restrict__ wkT,
                            const float* __restrict__ bq, const float* __restrict__ bk,
                            _Float16* __restrict__ q16, _Float16* __restrict__ k16) {
    const int lane  = threadIdx.x & 31;
    const int wv    = threadIdx.x >> 5;
    const int mtile = blockIdx.y * 4 + wv;              // 0..1023
    const int etile = blockIdx.x;                       // 0..15
    const _Float16* wT   = (blockIdx.z == 0) ? wqT : wkT;
    const float*    bias = (blockIdx.z == 0) ? bq  : bk;
    _Float16*       out  = (blockIdx.z == 0) ? q16 : k16;

    const _Float16* A0 = x16 + (size_t)mtile * 16 * CDIM;
    const _Float16* B0 = wT  + (size_t)etile * 16 * CDIM;

    v8f acc = {};
#pragma unroll
    for (int kk = 0; kk < CDIM; kk += 32) {
        v16h a = load_frag16(A0 + kk, CDIM, lane);
        v16h b = load_frag16(B0 + kk, CDIM, lane);
        acc = __builtin_amdgcn_wmma_f32_16x16x32_f16(false, a, false, b,
                                                     (short)0, acc, false, false);
    }
    // C/D layout: VGPR r, lane L -> M = r + 8*(L>>4), N = L&15
    const int n  = lane & 15;
    const int mb = (lane >> 4) * 8;
    const int e  = etile * 16 + n;
    const float bb = bias[e];
#pragma unroll
    for (int r = 0; r < 8; ++r) {
        out[(size_t)(mtile * 16 + mb + r) * CDIM + e] = (_Float16)(acc[r] + bb);
    }
}

// ---------------- scores = LeakyReLU(q @ k^T), written straight into edge_weight ----------------
// grid = (N/16, N/16/4, B), block = 128 (4 waves, one 16x16 tile each)
__global__ void scores_kernel(const _Float16* __restrict__ q16,
                              const _Float16* __restrict__ k16,
                              float* __restrict__ ew) {
    const int lane  = threadIdx.x & 31;
    const int wv    = threadIdx.x >> 5;
    const int jtile = blockIdx.x;                        // 0..127
    const int itile = blockIdx.y * 4 + wv;               // 0..127
    const int b     = blockIdx.z;                        // 0..7

    const _Float16* A0 = q16 + ((size_t)b * NNODE + itile * 16) * CDIM;
    const _Float16* B0 = k16 + ((size_t)b * NNODE + jtile * 16) * CDIM;  // B[e,j] = k[j,e]

    v8f acc = {};
#pragma unroll
    for (int kk = 0; kk < CDIM; kk += 32) {
        v16h a = load_frag16(A0 + kk, CDIM, lane);
        v16h bfr = load_frag16(B0 + kk, CDIM, lane);
        acc = __builtin_amdgcn_wmma_f32_16x16x32_f16(false, a, false, bfr,
                                                     (short)0, acc, false, false);
    }
    const int n  = lane & 15;
    const int mb = (lane >> 4) * 8;
#pragma unroll
    for (int r = 0; r < 8; ++r) {
        float s = acc[r];
        s = (s >= 0.f) ? s : SLOPE * s;                  // LeakyReLU
        size_t row = (size_t)b * NNODE + itile * 16 + mb + r;
        ew[row * NNODE + jtile * 16 + n] = s;
    }
}

// ---------------- top-K + softmax (in place on ew) + per-row loss contribution ----------------
// grid = ROWS blocks of 256 threads; row scores live in LDS (8 KB << 320 KB/WGP)
__global__ void topk_softmax_kernel(const float* __restrict__ x,
                                    const float* __restrict__ Sb,
                                    float* __restrict__ ew,
                                    float* __restrict__ contrib) {
    __shared__ float sv[NNODE];
    __shared__ float redv[256];
    __shared__ int   redi[256];
    __shared__ float topv[TOPK];
    __shared__ int   topi[TOPK];
    __shared__ float wts[TOPK];
    __shared__ float dsh;

    const int c = threadIdx.x;               // 0..255 (== channel index)
    const int r = blockIdx.x;                // 0..16383
    const int b = r >> 11;
    float* row = ew + (size_t)r * NNODE;

    for (int u = 0; u < NNODE / 256; ++u) sv[u * 256 + c] = row[u * 256 + c];
    const float xi = x[(size_t)r * CDIM + c];
    __syncthreads();

    // iterative top-32 (stable: smaller index wins ties, matching lax.top_k)
    for (int t = 0; t < TOPK; ++t) {
        float bv = -3.4e38f; int bi = 0;
        for (int u = 0; u < NNODE / 256; ++u) {
            int j = u * 256 + c;
            float v = sv[j];
            if (v > bv || (v == bv && j < bi)) { bv = v; bi = j; }
        }
        redv[c] = bv; redi[c] = bi;
        __syncthreads();
        for (int s = 128; s > 0; s >>= 1) {
            if (c < s) {
                float ov = redv[c + s]; int oi = redi[c + s];
                if (ov > redv[c] || (ov == redv[c] && oi < redi[c])) { redv[c] = ov; redi[c] = oi; }
            }
            __syncthreads();
        }
        if (c == 0) { topv[t] = redv[0]; topi[t] = redi[0]; sv[redi[0]] = -3.4e38f; }
        __syncthreads();
    }

    // softmax over [zeros with K values scattered]: denom = (N-K)*exp(0) + sum exp(v)
    if (c == 0) {
        float d = (float)(NNODE - TOPK);
        for (int t = 0; t < TOPK; ++t) d += expf(topv[t]);
        dsh = d;
    }
    __syncthreads();
    const float denom = dsh;
    const float base  = 1.0f / denom;
    if (c < TOPK) wts[c] = (expf(topv[c]) - 1.0f) * base;

    // rebuild normalized weights in LDS, then one coalesced writeback
    for (int u = 0; u < NNODE / 256; ++u) sv[u * 256 + c] = base;
    __syncthreads();
    if (c < TOPK) sv[topi[c]] = expf(topv[c]) * base;
    __syncthreads();
    for (int u = 0; u < NNODE / 256; ++u) row[u * 256 + c] = sv[u * 256 + c];

    // loss: contrib_i = ||x_i||^2 - x_i . ( Sb/denom + sum_t wts[t]*x_{topi[t]} )
    float y = Sb[b * CDIM + c] * base;
    for (int t = 0; t < TOPK; ++t)
        y += wts[t] * x[((size_t)b * NNODE + topi[t]) * CDIM + c];
    redv[c] = xi * xi - xi * y;
    __syncthreads();
    for (int s = 128; s > 0; s >>= 1) { if (c < s) redv[c] += redv[c + s]; __syncthreads(); }
    if (c == 0) contrib[r] = redv[0];
}

// ---------------- deterministic loss reduction ----------------
__global__ void loss_reduce_kernel(const float* __restrict__ contrib, float* __restrict__ loss) {
    __shared__ float red[256];
    int c = threadIdx.x;
    float s = 0.f;
    for (int k = 0; k < ROWS / 256; ++k) s += contrib[k * 256 + c];   // fixed order
    red[c] = s;
    __syncthreads();
    for (int st = 128; st > 0; st >>= 1) { if (c < st) red[c] += red[c + st]; __syncthreads(); }
    if (c == 0) *loss = red[0] / (float)BATCH;                        // mean over batch
}

// ---------------- edge_index fill: rows = off+i, cols = off+j (as float values) ----------------
__global__ void edge_index_kernel(float* __restrict__ ei0, float* __restrict__ ei1) {
    size_t t = (size_t)blockIdx.x * blockDim.x + threadIdx.x;
    size_t p = t * 4;                       // 4 consecutive j per thread
    if (p >= BNN) return;
    int j = (int)(p & (NNODE - 1));
    int i = (int)((p >> 11) & (NNODE - 1));
    int b = (int)(p >> 22);
    float rv = (float)(b * NNODE + i);
    float cb = (float)(b * NNODE + j);
    float4 r4 = make_float4(rv, rv, rv, rv);
    float4 c4 = make_float4(cb, cb + 1.f, cb + 2.f, cb + 3.f);
    ((float4*)ei0)[t] = r4;
    ((float4*)ei1)[t] = c4;
}

// ---------------- launcher ----------------
extern "C" void kernel_launch(void* const* d_in, const int* in_sizes, int n_in,
                              void* d_out, int out_size, void* d_ws, size_t ws_size,
                              hipStream_t stream) {
    const float* x  = (const float*)d_in[0];   // [16384, 256]
    const float* Wq = (const float*)d_in[1];   // [256, 256]
    const float* bq = (const float*)d_in[2];   // [256]
    const float* Wk = (const float*)d_in[3];   // [256, 256]
    const float* bk = (const float*)d_in[4];   // [256]

    float* out = (float*)d_out;
    float* ei0 = out;                  // [BNN]
    float* ei1 = out + BNN;            // [BNN]
    float* ew  = out + 2 * BNN;        // [BNN]  (scores written here, softmaxed in place)
    float* lossp = out + 3 * BNN;      // [1]

    // workspace layout (~25 MB)
    _Float16* x16 = (_Float16*)d_ws;                         // 16384*256
    _Float16* wqT = x16 + (size_t)ROWS * CDIM;               // 256*256
    _Float16* wkT = wqT + CDIM * CDIM;
    _Float16* q16 = wkT + CDIM * CDIM;                       // 16384*256
    _Float16* k16 = q16 + (size_t)ROWS * CDIM;
    float* Sb      = (float*)(k16 + (size_t)ROWS * CDIM);    // [8, 256]
    float* contrib = Sb + BATCH * CDIM;                      // [16384]

    // 1) prep: converts + transposes + column sums
    cvt_x_kernel<<<(ROWS * CDIM) / 256, 256, 0, stream>>>(x, x16, ROWS * CDIM);
    cvt_wT_kernel<<<(2 * CDIM * CDIM) / 256, 256, 0, stream>>>(Wq, Wk, wqT, wkT);
    colsum_kernel<<<BATCH, CDIM, 0, stream>>>(x, Sb);

    // 2) q/k projections (WMMA)
    proj_kernel<<<dim3(CDIM / 16, ROWS / 16 / 4, 2), 128, 0, stream>>>(
        x16, wqT, wkT, bq, bk, q16, k16);

    // 3) scores = LeakyReLU(q k^T) -> edge_weight region (WMMA)
    scores_kernel<<<dim3(NNODE / 16, NNODE / 16 / 4, BATCH), 128, 0, stream>>>(q16, k16, ew);

    // 4) top-K + softmax in place + per-row loss contributions
    topk_softmax_kernel<<<ROWS, 256, 0, stream>>>(x, Sb, ew, contrib);

    // 5) deterministic loss reduction
    loss_reduce_kernel<<<1, 256, 0, stream>>>(contrib, lossp);

    // 6) edge_index (pure arithmetic, float4 stores)
    edge_index_kernel<<<(unsigned)((BNN / 4 + 255) / 256), 256, 0, stream>>>(ei0, ei1);
}
